// RevisionDecoder_53506702574125
// MI455X (gfx1250) — matrix operand order repaired and checked by reading, not compiled, not run on hardware
//
#include <hip/hip_runtime.h>
#include <hip/hip_bf16.h>
#include <math.h>

#define DEV static __device__ __forceinline__

typedef _Float16 v16h __attribute__((ext_vector_type(16)));
typedef _Float16 v8h  __attribute__((ext_vector_type(8)));
typedef float    v8f  __attribute__((ext_vector_type(8)));

static constexpr int Bc = 16, Nc = 32, Kc = 8, Hc = 128, DEPTHc = 2;
static constexpr int NRBF = 16, VOC = 21, PDIM = 16;
static constexpr float LN1E4_OVER_P = 0.57564627324851142f; // ln(10000)/POS_DIM

DEV float sgnf(float x){ return (x > 0.f) ? 1.f : ((x < 0.f) ? -1.f : 0.f); }

// ---------------------------------------------------------------------------
// Weight pre-pack: f32 [Kreal, Hc] row-major  ->  f16 B-fragment order,
// zero-padded to Kp rows.  Layout: P[(((nt*(Kp/32) + c)*32 + lane)*16) + e]
// holds W[k][n] with k = c*32 + (lane>>4)*16 + e, n = nt*16 + (lane&15).
// ---------------------------------------------------------------------------
__global__ void __launch_bounds__(256)
k_packW(const float* __restrict__ W, int Kreal, int Kp, _Float16* __restrict__ P)
{
  int idx = blockIdx.x*256 + threadIdx.x;          // over Kp*Hc fragment elems
  if (idx >= Kp*Hc) return;
  int e    = idx & 15;
  int lane = (idx >> 4) & 31;
  int rest = idx >> 9;                             // nt*(Kp/32) + c
  int nchunks = Kp >> 5;
  int c  = rest % nchunks;
  int nt = rest / nchunks;
  int k = c*32 + (lane >> 4)*16 + e;
  int n = nt*16 + (lane & 15);
  P[idx] = (_Float16)((k < Kreal) ? W[(size_t)k*Hc + n] : 0.f);
}

// ---------------------------------------------------------------------------
// WMMA GEMM: out[M,Hc] = relu( A_f16[M,KP] @ Wpk + bias ).  One wave per
// 16x16 C tile; KP fully unrolled; pure b128-load + v_wmma inner stream.
// ---------------------------------------------------------------------------
template<int KP>
__global__ void __launch_bounds__(32)
k_gemm(const _Float16* __restrict__ A, const _Float16* __restrict__ Bp,
       const float* __restrict__ bias, float* __restrict__ out)
{
  const int lane = threadIdx.x;
  const int mt = blockIdx.x, nt = blockIdx.y;
  const int rlo = lane & 15;      // row (A) / col (B,C)
  const int kh  = lane >> 4;      // K-half selector
  const _Float16* Arow = A + (size_t)(mt*16 + rlo) * KP;
  const _Float16* Bt   = Bp + ((size_t)nt*(KP/32)*32 + lane) * 16;
  v8f acc = {0.f,0.f,0.f,0.f,0.f,0.f,0.f,0.f};
  #pragma unroll
  for (int c = 0; c < KP/32; c++) {
    v8h a0 = *(const v8h*)(Arow + c*32 + kh*16);
    v8h a1 = *(const v8h*)(Arow + c*32 + kh*16 + 8);
    v8h b0 = *(const v8h*)(Bt + (size_t)c*32*16);
    v8h b1 = *(const v8h*)(Bt + (size_t)c*32*16 + 8);
    v16h av, bv;
    #pragma unroll
    for (int e = 0; e < 8; e++) {
      av[e] = a0[e]; av[e+8] = a1[e];
      bv[e] = b0[e]; bv[e+8] = b1[e];
    }
    acc = __builtin_amdgcn_wmma_f32_16x16x32_f16(false, av, false, bv,
                                                 (short)0, acc, false, false);
  }
  const int ncol = nt*16 + rlo;
  float bb = bias[ncol];
  #pragma unroll
  for (int v = 0; v < 8; v++) {
    float x = fmaxf(acc[v] + bb, 0.f);
    int m = mt*16 + v + 8*kh;
    out[(size_t)m*Hc + ncol] = x;
  }
}

// ---------------------------------------------------------------------------
// Pack kernels (fp32 -> f16 A-matrices, with concat / gather fused in)
// ---------------------------------------------------------------------------
__global__ void __launch_bounds__(256)
k_packnode(const float* __restrict__ V, int node_in,
           const float* __restrict__ emb, const int* __restrict__ S,
           _Float16* __restrict__ A, int Kp)
{
  int row = blockIdx.x;                     // 0..B*N-1
  int si = S[row];
  for (int j = threadIdx.x; j < Kp; j += blockDim.x) {
    float v;
    if (j < node_in)            v = V[(size_t)row*node_in + j];
    else if (j < node_in + Hc)  v = emb[(size_t)si*Hc + (j - node_in)];
    else                        v = 0.f;
    A[(size_t)row*Kp + j] = (_Float16)v;
  }
}

__global__ void __launch_bounds__(64)
k_packedge(const float* __restrict__ E, int edge_in,
           _Float16* __restrict__ A, int Kp)
{
  int row = blockIdx.x;                     // 0..B*N*K-1
  for (int j = threadIdx.x; j < Kp; j += blockDim.x) {
    float v = (j < edge_in) ? E[(size_t)row*edge_in + j] : 0.f;
    A[(size_t)row*Kp + j] = (_Float16)v;
  }
}

__global__ void __launch_bounds__(256)
k_packmsg(const float* __restrict__ he, const float* __restrict__ h,
          const int* __restrict__ Eix, _Float16* __restrict__ A)
{
  int row = blockIdx.x;                     // 0..B*N*K-1
  int b = row / (Nc*Kc);
  int nb = Eix[row];
  const float* hr = h + ((size_t)b*Nc + nb)*Hc;
  for (int j = threadIdx.x; j < 2*Hc; j += blockDim.x) {
    float v = (j < Hc) ? he[(size_t)row*Hc + j] : hr[j - Hc];
    A[(size_t)row*2*Hc + j] = (_Float16)v;
  }
}

// agg = mean_k(msg * nei_mask);  A = [h0, agg] (f16)
__global__ void __launch_bounds__(Hc)
k_aggpack(const float* __restrict__ msg, const float* __restrict__ h0,
          const int* __restrict__ Eix, const float* __restrict__ mask,
          _Float16* __restrict__ A)
{
  int row = blockIdx.x;                     // node row 0..B*N-1
  int b = row / Nc;
  int j = threadIdx.x;
  float s = 0.f;
  #pragma unroll
  for (int k = 0; k < Kc; k++) {
    int e = row*Kc + k;
    float m = mask[b*Nc + Eix[e]];
    s += msg[(size_t)e*Hc + j] * m;
  }
  s *= (1.f / Kc);
  A[(size_t)row*2*Hc + j]      = (_Float16)h0[(size_t)row*Hc + j];
  A[(size_t)row*2*Hc + Hc + j] = (_Float16)s;
}

__global__ void __launch_bounds__(Hc)
k_maskh(float* __restrict__ h, const float* __restrict__ mask)
{
  int row = blockIdx.x;
  h[(size_t)row*Hc + threadIdx.x] *= mask[row];
}

// ---------------------------------------------------------------------------
// Geometric feature kernels (fp32 VALU, one block per batch)
// ---------------------------------------------------------------------------
__global__ void __launch_bounds__(128)
k_dihedrals(const float* __restrict__ X, float* __restrict__ Vout)
{
  __shared__ float U[95][3];
  __shared__ float Dang[96];
  int b = blockIdx.x, t = threadIdx.x;
  if (t < 95) {
    int n0 = t/3, a0 = t%3, n1 = (t+1)/3, a1 = (t+1)%3;
    const float* p0 = X + (((size_t)b*Nc + n0)*4 + a0)*3;
    const float* p1 = X + (((size_t)b*Nc + n1)*4 + a1)*3;
    float dx = p1[0]-p0[0], dy = p1[1]-p0[1], dz = p1[2]-p0[2];
    float nr = sqrtf(dx*dx + dy*dy + dz*dz) + 1e-8f;
    U[t][0] = dx/nr; U[t][1] = dy/nr; U[t][2] = dz/nr;
  }
  if (t < 96) Dang[t] = 0.f;
  __syncthreads();
  if (t < 93) {
    float u2x=U[t][0],   u2y=U[t][1],   u2z=U[t][2];
    float u1x=U[t+1][0], u1y=U[t+1][1], u1z=U[t+1][2];
    float u0x=U[t+2][0], u0y=U[t+2][1], u0z=U[t+2][2];
    float c2x = u2y*u1z - u2z*u1y, c2y = u2z*u1x - u2x*u1z, c2z = u2x*u1y - u2y*u1x;
    float n2n = sqrtf(c2x*c2x + c2y*c2y + c2z*c2z) + 1e-8f;
    c2x/=n2n; c2y/=n2n; c2z/=n2n;
    float c1x = u1y*u0z - u1z*u0y, c1y = u1z*u0x - u1x*u0z, c1z = u1x*u0y - u1y*u0x;
    float n1n = sqrtf(c1x*c1x + c1y*c1y + c1z*c1z) + 1e-8f;
    c1x/=n1n; c1y/=n1n; c1z/=n1n;
    float cosD = c2x*c1x + c2y*c1y + c2z*c1z;
    cosD = fminf(fmaxf(cosD, -1.f + 1e-7f), 1.f - 1e-7f);
    float sg = u2x*c1x + u2y*c1y + u2z*c1z;
    Dang[t+1] = sgnf(sg) * acosf(cosD);   // pad (1,2)
  }
  __syncthreads();
  if (t < 96) {
    int n = t/3, c = t%3;
    Vout[((size_t)b*Nc + n)*6 + c]     = cosf(Dang[t]);
    Vout[((size_t)b*Nc + n)*6 + 3 + c] = sinf(Dang[t]);
  }
}

__global__ void __launch_bounds__(32)
k_adfeat(const float* __restrict__ X, float* __restrict__ AD)
{
  __shared__ float U[31][3];
  int b = blockIdx.x, t = threadIdx.x;
  if (t < 31) {
    const float* p0 = X + (((size_t)b*Nc + t  )*4 + 1)*3;
    const float* p1 = X + (((size_t)b*Nc + t+1)*4 + 1)*3;
    float dx = p1[0]-p0[0], dy = p1[1]-p0[1], dz = p1[2]-p0[2];
    float nr = sqrtf(dx*dx + dy*dy + dz*dz) + 1e-8f;
    U[t][0] = dx/nr; U[t][1] = dy/nr; U[t][2] = dz/nr;
  }
  __syncthreads();
  float o0 = 0.f, o1 = 0.f, o2 = 0.f;
  if (t >= 1 && t < 30) {
    int m = t - 1;
    float u2x=U[m][0],   u2y=U[m][1],   u2z=U[m][2];
    float u1x=U[m+1][0], u1y=U[m+1][1], u1z=U[m+1][2];
    float u0x=U[m+2][0], u0y=U[m+2][1], u0z=U[m+2][2];
    float cosA = -(u1x*u0x + u1y*u0y + u1z*u0z);
    cosA = fminf(fmaxf(cosA, -1.f + 1e-6f), 1.f - 1e-6f);
    float A = acosf(cosA);
    float c2x = u2y*u1z - u2z*u1y, c2y = u2z*u1x - u2x*u1z, c2z = u2x*u1y - u2y*u1x;
    float n2n = sqrtf(c2x*c2x + c2y*c2y + c2z*c2z) + 1e-8f;
    c2x/=n2n; c2y/=n2n; c2z/=n2n;
    float c1x = u1y*u0z - u1z*u0y, c1y = u1z*u0x - u1x*u0z, c1z = u1x*u0y - u1y*u0x;
    float n1n = sqrtf(c1x*c1x + c1y*c1y + c1z*c1z) + 1e-8f;
    c1x/=n1n; c1y/=n1n; c1z/=n1n;
    float cosDa = c2x*c1x + c2y*c1y + c2z*c1z;
    cosDa = fminf(fmaxf(cosDa, -1.f + 1e-6f), 1.f - 1e-6f);
    float Da = sgnf(u2x*c1x + u2y*c1y + u2z*c1z) * acosf(cosDa);
    o0 = cosf(A); o1 = sinf(A)*cosf(Da); o2 = sinf(A)*sinf(Da);
  }
  if (t < 32) {
    AD[((size_t)b*Nc + t)*3 + 0] = o0;
    AD[((size_t)b*Nc + t)*3 + 1] = o1;
    AD[((size_t)b*Nc + t)*3 + 2] = o2;
  }
}

__global__ void __launch_bounds__(1024)
k_trueD(const float* __restrict__ X, const float* __restrict__ mask,
        float* __restrict__ TD)
{
  __shared__ float Xca[32][3];
  int b = blockIdx.x, t = threadIdx.x;
  if (t < 32) {
    const float* p = X + (((size_t)b*Nc + t)*4 + 1)*3;
    Xca[t][0] = p[0]; Xca[t][1] = p[1]; Xca[t][2] = p[2];
  }
  __syncthreads();
  int i = t >> 5, j = t & 31;
  float dx = Xca[i][0]-Xca[j][0], dy = Xca[i][1]-Xca[j][1], dz = Xca[i][2]-Xca[j][2];
  float m2 = mask[b*Nc + i] * mask[b*Nc + j];
  float v = m2 * sqrtf(dx*dx + dy*dy + dz*dz + 1e-6f);
  TD[((size_t)b*Nc + i)*Nc + j] = v*v;
}

__global__ void __launch_bounds__(1024)
k_dloss(const float* __restrict__ X, const float* __restrict__ TD,
        float* __restrict__ dl)
{
  __shared__ float Xca[32][3];
  int b = blockIdx.x, t = threadIdx.x;
  if (t < 32) {
    const float* p = X + (((size_t)b*Nc + t)*4 + 1)*3;
    Xca[t][0] = p[0]; Xca[t][1] = p[1]; Xca[t][2] = p[2];
  }
  __syncthreads();
  int i = t >> 5, j = t & 31;
  float dx = Xca[i][0]-Xca[j][0], dy = Xca[i][1]-Xca[j][1], dz = Xca[i][2]-Xca[j][2];
  float d2 = dx*dx + dy*dy + dz*dz;
  size_t idx = ((size_t)b*Nc + i)*Nc + j;
  float ad = fabsf(d2 - TD[idx]);
  dl[idx] += (ad < 1.f) ? 0.5f*ad*ad : ad - 0.5f;
}

// Full protein_features edge path: D, topK, posenc, rbf, orientations+quaternions
__global__ void __launch_bounds__(256)
k_protfeat(const float* __restrict__ X, const float* __restrict__ mask,
           int* __restrict__ Eidx, float* __restrict__ E)
{
  __shared__ float Xca[32][3];
  __shared__ float Dm[32][32];
  __shared__ float rmax[32];
  __shared__ float O[32][9];
  __shared__ float U[31][3];
  __shared__ int   eix[32][8];
  __shared__ float edv[32][8];
  int b = blockIdx.x, t = threadIdx.x;
  if (t < 32) {
    const float* p = X + (((size_t)b*Nc + t)*4 + 1)*3;
    Xca[t][0] = p[0]; Xca[t][1] = p[1]; Xca[t][2] = p[2];
  }
  __syncthreads();
  for (int idx = t; idx < 1024; idx += 256) {
    int i = idx >> 5, j = idx & 31;
    float dx = Xca[i][0]-Xca[j][0], dy = Xca[i][1]-Xca[j][1], dz = Xca[i][2]-Xca[j][2];
    float m2 = mask[b*Nc + i] * mask[b*Nc + j];
    Dm[i][j] = m2 * sqrtf(dx*dx + dy*dy + dz*dz + 1e-6f);
  }
  __syncthreads();
  if (t < 32) {
    float mx = Dm[t][0];
    for (int j = 1; j < 32; j++) mx = fmaxf(mx, Dm[t][j]);
    rmax[t] = mx;
  }
  __syncthreads();
  for (int idx = t; idx < 1024; idx += 256) {
    int i = idx >> 5, j = idx & 31;
    float m2 = mask[b*Nc + i] * mask[b*Nc + j];
    Dm[i][j] += (1.f - m2) * rmax[i];
  }
  __syncthreads();
  if (t < 32) {           // stable top-K smallest (ties -> smallest index)
    unsigned used = 0u;
    for (int k = 0; k < Kc; k++) {
      int best = 0; float bv = 3.4e38f;
      for (int j = 0; j < 32; j++)
        if (!((used >> j) & 1u) && Dm[t][j] < bv) { bv = Dm[t][j]; best = j; }
      used |= (1u << best);
      eix[t][k] = best; edv[t][k] = bv;
    }
  }
  __syncthreads();
  if (t < 31) {
    float dx = Xca[t+1][0]-Xca[t][0], dy = Xca[t+1][1]-Xca[t][1], dz = Xca[t+1][2]-Xca[t][2];
    float nr = sqrtf(dx*dx + dy*dy + dz*dz) + 1e-8f;
    U[t][0] = dx/nr; U[t][1] = dy/nr; U[t][2] = dz/nr;
  }
  __syncthreads();
  if (t < 32) {           // O frames, padded (1,2): rows 1..29
    if (t >= 1 && t <= 29) {
      int m = t - 1;
      float u2x=U[m][0],   u2y=U[m][1],   u2z=U[m][2];
      float u1x=U[m+1][0], u1y=U[m+1][1], u1z=U[m+1][2];
      float cx = u2y*u1z - u2z*u1y, cy = u2z*u1x - u2x*u1z, cz = u2x*u1y - u2y*u1x;
      float nn = sqrtf(cx*cx + cy*cy + cz*cz) + 1e-8f;
      float n2x = cx/nn, n2y = cy/nn, n2z = cz/nn;
      float ox = u2x-u1x, oy = u2y-u1y, oz = u2z-u1z;
      float on = sqrtf(ox*ox + oy*oy + oz*oz) + 1e-8f;
      ox/=on; oy/=on; oz/=on;
      float wx = oy*n2z - oz*n2y, wy = oz*n2x - ox*n2z, wz = ox*n2y - oy*n2x;
      O[t][0]=ox; O[t][1]=oy; O[t][2]=oz;
      O[t][3]=n2x; O[t][4]=n2y; O[t][5]=n2z;
      O[t][6]=wx; O[t][7]=wy; O[t][8]=wz;
    } else {
      #pragma unroll
      for (int q = 0; q < 9; q++) O[t][q] = 0.f;
    }
  }
  __syncthreads();
  { // one thread per (n,k) : 256 threads exactly
    int n = t >> 3, k = t & 7;
    int j = eix[n][k];
    Eidx[((size_t)b*Nc + n)*Kc + k] = j;
    float* e = E + (((size_t)b*Nc + n)*Kc + k)*39;
    float d = (float)(j - n);
    #pragma unroll
    for (int p = 0; p < 8; p++) {
      float fr = expf(-(float)(2*p) * LN1E4_OVER_P);
      float ang = d * fr;
      e[p]   = cosf(ang);
      e[8+p] = sinf(ang);
    }
    float Dv = edv[n][k];
    #pragma unroll
    for (int r = 0; r < NRBF; r++) {
      float mu = (20.f/15.f) * (float)r;
      float z = (Dv - mu) * (1.f/1.25f);
      e[16+r] = expf(-z*z);
    }
    float dX0 = Xca[j][0]-Xca[n][0], dX1 = Xca[j][1]-Xca[n][1], dX2 = Xca[j][2]-Xca[n][2];
    float du0 = O[n][0]*dX0 + O[n][1]*dX1 + O[n][2]*dX2;
    float du1 = O[n][3]*dX0 + O[n][4]*dX1 + O[n][5]*dX2;
    float du2 = O[n][6]*dX0 + O[n][7]*dX1 + O[n][8]*dX2;
    float dn = sqrtf(du0*du0 + du1*du1 + du2*du2) + 1e-8f;
    du0/=dn; du1/=dn; du2/=dn;
    float R[3][3];
    #pragma unroll
    for (int i = 0; i < 3; i++)
      #pragma unroll
      for (int l = 0; l < 3; l++)
        R[i][l] = O[n][i]*O[j][l] + O[n][3+i]*O[j][3+l] + O[n][6+i]*O[j][6+l];
    float Rxx = R[0][0], Ryy = R[1][1], Rzz = R[2][2];
    float m0 = 0.5f*sqrtf(fabsf(1.f + Rxx - Ryy - Rzz));
    float m1 = 0.5f*sqrtf(fabsf(1.f - Rxx + Ryy - Rzz));
    float m2 = 0.5f*sqrtf(fabsf(1.f - Rxx - Ryy + Rzz));
    float q0 = sgnf(R[2][1]-R[1][2]) * m0;
    float q1 = sgnf(R[0][2]-R[2][0]) * m1;
    float q2 = sgnf(R[1][0]-R[0][1]) * m2;
    float w  = 0.5f*sqrtf(fmaxf(0.f, 1.f + Rxx + Ryy + Rzz));
    float qn = sqrtf(q0*q0 + q1*q1 + q2*q2 + w*w) + 1e-8f;
    e[32]=du0; e[33]=du1; e[34]=du2;
    e[35]=q0/qn; e[36]=q1/qn; e[37]=q2/qn; e[38]=w/qn;
  }
}

// init_coords features: Vn, E_idx (|i-j| nearest, stable), E=[posenc, rbf(3*d)]
__global__ void __launch_bounds__(256)
k_initfeat(float* __restrict__ Vn, float* __restrict__ Ei, int* __restrict__ EidxI)
{
  __shared__ int eix[32][8];
  __shared__ int edd[32][8];
  int b = blockIdx.x, t = threadIdx.x;
  if (t < 32) {
    int tmp[8]; int cnt = 0;
    tmp[cnt++] = t;
    for (int d = 1; cnt < 8; d++) {
      if (t - d >= 0 && cnt < 8) tmp[cnt++] = t - d;
      if (t + d < 32 && cnt < 8) tmp[cnt++] = t + d;
    }
    #pragma unroll
    for (int k = 0; k < 8; k++) {
      eix[t][k] = tmp[k];
      int dd = tmp[k] - t; edd[t][k] = dd < 0 ? -dd : dd;
    }
  }
  __syncthreads();
  for (int idx = t; idx < 32*PDIM; idx += 256) {
    int n = idx >> 4, c = idx & 15;
    int p = c & 7;
    float fr = expf(-(float)(2*p) * LN1E4_OVER_P);
    float ang = (float)n * fr;
    Vn[((size_t)b*Nc + n)*PDIM + c] = (c < 8) ? cosf(ang) : sinf(ang);
  }
  {
    int n = t >> 3, k = t & 7;
    int j = eix[n][k];
    EidxI[((size_t)b*Nc + n)*Kc + k] = j;
    float* e = Ei + (((size_t)b*Nc + n)*Kc + k)*32;
    float d = (float)(j - n);
    #pragma unroll
    for (int p = 0; p < 8; p++) {
      float fr = expf(-(float)(2*p) * LN1E4_OVER_P);
      float ang = d * fr;
      e[p] = cosf(ang); e[8+p] = sinf(ang);
    }
    float Dv = 3.f * (float)edd[n][k];
    #pragma unroll
    for (int r = 0; r < NRBF; r++) {
      float mu = (20.f/15.f) * (float)r;
      float z = (Dv - mu) * (1.f/1.25f);
      e[16+r] = expf(-z*z);
    }
  }
}

// ---------------------------------------------------------------------------
// Output heads & loss kernels
// ---------------------------------------------------------------------------
__global__ void __launch_bounds__(512)
k_logits(const float* __restrict__ h, const float* __restrict__ Ws,
         const float* __restrict__ bs, const int* __restrict__ trueS,
         const float* __restrict__ mask, int* __restrict__ S,
         float* __restrict__ sloss, int t)
{
  __shared__ float lg[16][22];
  __shared__ float pen[16];
  int bb = threadIdx.x >> 5, v = threadIdx.x & 31;
  if (v < VOC) {
    float s = bs[v];
    const float* hr = h + ((size_t)bb*Nc + t)*Hc;
    for (int j = 0; j < Hc; j++) s += hr[j]*Ws[j*VOC + v];
    lg[bb][v] = s;
  }
  __syncthreads();
  if (v == 0) {
    float mx = lg[bb][0];
    for (int c = 1; c < VOC; c++) mx = fmaxf(mx, lg[bb][c]);
    float se = 0.f;
    for (int c = 0; c < VOC; c++) se += expf(lg[bb][c] - mx);
    float lse = mx + logf(se);
    int tgt = trueS[bb*Nc + t];
    pen[bb] = (lse - lg[bb][tgt]) * mask[bb*Nc + t];
    S[bb*Nc + t] = tgt;
  }
  __syncthreads();
  if (threadIdx.x == 0) {
    float s = 0.f;
    for (int i = 0; i < 16; i++) s += pen[i];
    sloss[0] += s;              // single-thread: deterministic
  }
}

__global__ void __launch_bounds__(128)
k_predict(const float* __restrict__ h, const float* __restrict__ W,
          const float* __restrict__ bias, float* __restrict__ Xout)
{
  int idx = blockIdx.x*128 + threadIdx.x;
  if (idx >= Bc*Nc*12) return;
  int row = idx/12, c = idx%12;
  const float* hr = h + (size_t)row*Hc;
  float s = bias[c];
  for (int j = 0; j < Hc; j++) s += hr[j]*W[j*12 + c];
  Xout[idx] = s;
}

__global__ void k_sqacc(float* __restrict__ acc, const float* __restrict__ x,
                        const float* __restrict__ ref, int n)
{
  int i = blockIdx.x*blockDim.x + threadIdx.x;
  if (i < n) { float d = x[i] - ref[i]; acc[i] += d*d; }
}

DEV float block_sum(float* red, float x)
{
  int t = threadIdx.x;
  red[t] = x; __syncthreads();
  for (int w = 128; w > 0; w >>= 1) {
    if (t < w) red[t] += red[t + w];
    __syncthreads();
  }
  float r = red[0]; __syncthreads();
  return r;
}

__global__ void __launch_bounds__(256)
k_finalize(const float* __restrict__ dl, const float* __restrict__ vl,
           const float* __restrict__ al, const float* __restrict__ sl,
           const float* __restrict__ mask, float* __restrict__ out)
{
  __shared__ float red[256];
  int t = threadIdx.x;
  float s_m = 0.f, s_m2 = 0.f, s_d = 0.f, s_v = 0.f, s_a = 0.f;
  for (int i = t; i < Bc*Nc; i += 256) s_m += mask[i];
  for (int i = t; i < Bc*Nc*Nc; i += 256) {
    int b = i / (Nc*Nc); int r = i % (Nc*Nc); int ii = r / Nc, jj = r % Nc;
    float m2 = mask[b*Nc + ii] * mask[b*Nc + jj];
    s_m2 += m2; s_d += dl[i]*m2;
  }
  for (int i = t; i < Bc*Nc*6; i += 256) s_v += vl[i]*mask[i/6];
  for (int i = t; i < Bc*Nc*3; i += 256) s_a += al[i]*mask[i/3];
  float Sm  = block_sum(red, s_m);
  float Sm2 = block_sum(red, s_m2);
  float Sd  = block_sum(red, s_d);
  float Sv  = block_sum(red, s_v);
  float Sa  = block_sum(red, s_a);
  if (t == 0) out[0] = sl[0]/Sm + Sd/Sm2 + Sv/Sm + Sa/Sm;
}

// ---------------------------------------------------------------------------
// Host orchestration
// ---------------------------------------------------------------------------
extern "C" void kernel_launch(void* const* d_in, const int* in_sizes, int n_in,
                              void* d_out, int out_size, void* d_ws, size_t ws_size,
                              hipStream_t stream)
{
  (void)in_sizes; (void)n_in; (void)out_size; (void)ws_size;
  // JAX pytree (sorted-key) flattening:
  // 0:L 1:mask 2:O_d0_W 3:O_d0_b 4:O_d_W 5:O_d_b 6:O_s_W 7:O_s_b
  // 8..16:init{W_e,W_msg,W_upd,W_v,b_e,b_msg,b_upd,b_v,emb}
  // 17..25:seq{...} 26..34:struct{...} 35:true_S 36:true_X
  const float* mask = (const float*)d_in[1];
  const float* Od0W = (const float*)d_in[2];
  const float* Od0b = (const float*)d_in[3];
  const float* OdW  = (const float*)d_in[4];
  const float* Odb  = (const float*)d_in[5];
  const float* OsW  = (const float*)d_in[6];
  const float* Osb  = (const float*)d_in[7];
  struct Mpn { const float *W_e,*W_msg,*W_upd,*W_v,*b_e,*b_msg,*b_upd,*b_v,*emb; };
  auto mk = [&](int i) {
    Mpn p;
    p.W_e  =(const float*)d_in[i+0]; p.W_msg=(const float*)d_in[i+1];
    p.W_upd=(const float*)d_in[i+2]; p.W_v  =(const float*)d_in[i+3];
    p.b_e  =(const float*)d_in[i+4]; p.b_msg=(const float*)d_in[i+5];
    p.b_upd=(const float*)d_in[i+6]; p.b_v  =(const float*)d_in[i+7];
    p.emb  =(const float*)d_in[i+8];
    return p;
  };
  Mpn Pinit = mk(8), Pseq = mk(17), Pstr = mk(26);
  const int*   trueS = (const int*)d_in[35];
  const float* trueX = (const float*)d_in[36];

  char* wb = (char*)d_ws; size_t off = 0;
  auto alloc = [&](size_t bytes) -> void* {
    void* p = wb + off; off += (bytes + 255) & ~(size_t)255; return p;
  };
  float* trueD  = (float*)alloc((size_t)Bc*Nc*Nc*4);
  float* trueV  = (float*)alloc((size_t)Bc*Nc*6*4);
  float* trueAD = (float*)alloc((size_t)Bc*Nc*3*4);
  float* Vf     = (float*)alloc((size_t)Bc*Nc*6*4);
  float* Vtmp   = (float*)alloc((size_t)Bc*Nc*6*4);
  float* ADtmp  = (float*)alloc((size_t)Bc*Nc*3*4);
  float* Vn     = (float*)alloc((size_t)Bc*Nc*PDIM*4);
  float* Einit  = (float*)alloc((size_t)Bc*Nc*Kc*32*4);
  int*   EixI   = (int*)  alloc((size_t)Bc*Nc*Kc*4);
  float* Ebuf   = (float*)alloc((size_t)Bc*Nc*Kc*39*4);
  int*   Eix    = (int*)  alloc((size_t)Bc*Nc*Kc*4);
  int*   S      = (int*)  alloc((size_t)Bc*Nc*4);
  float* h0b    = (float*)alloc((size_t)Bc*Nc*Hc*4);
  float* heb    = (float*)alloc((size_t)Bc*Nc*Kc*Hc*4);
  float* hb     = (float*)alloc((size_t)Bc*Nc*Hc*4);
  float* msgb   = (float*)alloc((size_t)Bc*Nc*Kc*Hc*4);
  _Float16* A_node = (_Float16*)alloc((size_t)Bc*Nc*160*2);
  _Float16* A_edge = (_Float16*)alloc((size_t)Bc*Nc*Kc*64*2);
  _Float16* A_msg  = (_Float16*)alloc((size_t)Bc*Nc*Kc*2*Hc*2);
  _Float16* A_upd  = (_Float16*)alloc((size_t)Bc*Nc*2*Hc*2);
  float* X0     = (float*)alloc((size_t)Bc*Nc*12*4);
  float* X1     = (float*)alloc((size_t)Bc*Nc*12*4);
  float* dloss  = (float*)alloc((size_t)Bc*Nc*Nc*4);
  float* vloss  = (float*)alloc((size_t)Bc*Nc*6*4);
  float* aloss  = (float*)alloc((size_t)Bc*Nc*3*4);
  float* sloss  = (float*)alloc(4);

  hipMemsetAsync(dloss, 0, (size_t)Bc*Nc*Nc*4, stream);
  hipMemsetAsync(vloss, 0, (size_t)Bc*Nc*6*4, stream);
  hipMemsetAsync(aloss, 0, (size_t)Bc*Nc*3*4, stream);
  hipMemsetAsync(sloss, 0, 4, stream);
  hipMemsetAsync(S,     0, (size_t)Bc*Nc*4, stream);

  // --- pre-pack all GEMM weights into f16 B-fragment layout (once) ---
  struct PMpn {
    const float *b_v, *b_e, *b_msg, *b_upd, *emb;
    _Float16 *Wv, *We, *Wmsg[DEPTHc], *Wupd[DEPTHc];
    int node_in, edge_in, KpN, KpE;
  };
  auto prep = [&](const Mpn& P, int node_in, int edge_in) {
    PMpn q;
    q.b_v = P.b_v; q.b_e = P.b_e; q.b_msg = P.b_msg; q.b_upd = P.b_upd; q.emb = P.emb;
    q.node_in = node_in; q.edge_in = edge_in;
    q.KpN = ((node_in + Hc) + 31)/32*32;
    q.KpE = (edge_in + 31)/32*32;
    q.Wv = (_Float16*)alloc((size_t)q.KpN*Hc*2);
    q.We = (_Float16*)alloc((size_t)q.KpE*Hc*2);
    k_packW<<<(q.KpN*Hc + 255)/256, 256, 0, stream>>>(P.W_v, node_in + Hc, q.KpN, q.Wv);
    k_packW<<<(q.KpE*Hc + 255)/256, 256, 0, stream>>>(P.W_e, edge_in, q.KpE, q.We);
    for (int d = 0; d < DEPTHc; d++) {
      q.Wmsg[d] = (_Float16*)alloc((size_t)2*Hc*Hc*2);
      q.Wupd[d] = (_Float16*)alloc((size_t)2*Hc*Hc*2);
      k_packW<<<(2*Hc*Hc + 255)/256, 256, 0, stream>>>(P.W_msg + (size_t)d*2*Hc*Hc, 2*Hc, 2*Hc, q.Wmsg[d]);
      k_packW<<<(2*Hc*Hc + 255)/256, 256, 0, stream>>>(P.W_upd + (size_t)d*2*Hc*Hc, 2*Hc, 2*Hc, q.Wupd[d]);
    }
    return q;
  };
  PMpn Qinit = prep(Pinit, PDIM, PDIM + NRBF);
  PMpn Qseq  = prep(Pseq, 6, 39);
  PMpn Qstr  = prep(Pstr, 6, 39);

  auto launch_gemm = [&](const _Float16* A, int Kp, const _Float16* Bp,
                         const float* bias, float* out, int Mrows) {
    dim3 g(Mrows/16, Hc/16);
    switch (Kp) {
      case 32:  k_gemm<32> <<<g, 32, 0, stream>>>(A, Bp, bias, out); break;
      case 64:  k_gemm<64> <<<g, 32, 0, stream>>>(A, Bp, bias, out); break;
      case 160: k_gemm<160><<<g, 32, 0, stream>>>(A, Bp, bias, out); break;
      default:  k_gemm<256><<<g, 32, 0, stream>>>(A, Bp, bias, out); break;
    }
  };

  k_trueD<<<Bc, 1024, 0, stream>>>(trueX, mask, trueD);
  k_dihedrals<<<Bc, 128, 0, stream>>>(trueX, trueV);
  k_adfeat<<<Bc, 32, 0, stream>>>(trueX, trueAD);

  auto run_mpn = [&](const PMpn& Q, const float* V, const float* Eb,
                     const int* Ei, float* hout) {
    k_packnode<<<Bc*Nc, 256, 0, stream>>>(V, Q.node_in, Q.emb, S, A_node, Q.KpN);
    launch_gemm(A_node, Q.KpN, Q.Wv, Q.b_v, h0b, Bc*Nc);
    k_packedge<<<Bc*Nc*Kc, 64, 0, stream>>>(Eb, Q.edge_in, A_edge, Q.KpE);
    launch_gemm(A_edge, Q.KpE, Q.We, Q.b_e, heb, Bc*Nc*Kc);
    const float* hc = h0b;
    for (int d = 0; d < DEPTHc; d++) {
      k_packmsg<<<Bc*Nc*Kc, 256, 0, stream>>>(heb, hc, Ei, A_msg);
      launch_gemm(A_msg, 2*Hc, Q.Wmsg[d], Q.b_msg + d*Hc, msgb, Bc*Nc*Kc);
      k_aggpack<<<Bc*Nc, Hc, 0, stream>>>(msgb, h0b, Ei, mask, A_upd);
      launch_gemm(A_upd, 2*Hc, Q.Wupd[d], Q.b_upd + d*Hc, hout, Bc*Nc);
      hc = hout;
    }
    k_maskh<<<Bc*Nc, Hc, 0, stream>>>(hout, mask);
  };

  // --- init_coords + initial losses ---
  k_initfeat<<<Bc, 256, 0, stream>>>(Vn, Einit, EixI);
  run_mpn(Qinit, Vn, Einit, EixI, hb);
  k_predict<<<(Bc*Nc*12 + 127)/128, 128, 0, stream>>>(hb, Od0W, Od0b, X0);
  k_dloss<<<Bc, 1024, 0, stream>>>(X0, trueD, dloss);
  k_dihedrals<<<Bc, 128, 0, stream>>>(X0, Vtmp);
  k_sqacc<<<(Bc*Nc*6 + 255)/256, 256, 0, stream>>>(vloss, Vtmp, trueV, Bc*Nc*6);
  k_adfeat<<<Bc, 32, 0, stream>>>(X0, ADtmp);
  k_sqacc<<<(Bc*Nc*3 + 255)/256, 256, 0, stream>>>(aloss, ADtmp, trueAD, Bc*Nc*3);

  // --- autoregressive scan over t = 0..N-1 ---
  float* Xc = X0; float* Xn = X1;
  for (int t = 0; t < Nc; t++) {
    k_dihedrals<<<Bc, 128, 0, stream>>>(Xc, Vf);
    k_protfeat<<<Bc, 256, 0, stream>>>(Xc, mask, Eix, Ebuf);
    run_mpn(Qseq, Vf, Ebuf, Eix, hb);
    k_logits<<<1, 512, 0, stream>>>(hb, OsW, Osb, trueS, mask, S, sloss, t);
    run_mpn(Qstr, Vf, Ebuf, Eix, hb);
    k_predict<<<(Bc*Nc*12 + 127)/128, 128, 0, stream>>>(hb, OdW, Odb, Xn);
    k_dloss<<<Bc, 1024, 0, stream>>>(Xn, trueD, dloss);
    k_dihedrals<<<Bc, 128, 0, stream>>>(Xn, Vtmp);
    k_sqacc<<<(Bc*Nc*6 + 255)/256, 256, 0, stream>>>(vloss, Vtmp, trueV, Bc*Nc*6);
    k_adfeat<<<Bc, 32, 0, stream>>>(Xn, ADtmp);
    k_sqacc<<<(Bc*Nc*3 + 255)/256, 256, 0, stream>>>(aloss, ADtmp, trueAD, Bc*Nc*3);
    float* tmp = Xc; Xc = Xn; Xn = tmp;
  }

  k_finalize<<<1, 256, 0, stream>>>(dloss, vloss, aloss, sloss, mask, (float*)d_out);
}
